// DiscreteLatentDistributionVQ_78151224918431
// MI455X (gfx1250) — matrix operand order repaired and checked by reading, not compiled, not run on hardware
//
#include <hip/hip_runtime.h>
#include <hip/hip_bf16.h>
#include <stdint.h>

// ---------------------------------------------------------------------------
// Problem constants (from reference)
// ---------------------------------------------------------------------------
#define N_ROWS 65536
#define D_IN   1024
#define H_DIM  128
#define F_DIM  512
#define K_LAT  1024
#define INV_TEMP 10.0f   // 1 / TEMP

typedef __attribute__((ext_vector_type(16))) __bf16 v16bf;
typedef __attribute__((ext_vector_type(8)))  float  v8f;

// Hardware bf16 conversion (v_cvt_bf16_f32 / v_cvt_pk_bf16_f32), RNE.
__device__ __forceinline__ unsigned short f2bfh(float f) {
    __bf16 b = (__bf16)f;
    return __builtin_bit_cast(unsigned short, b);
}
__device__ __forceinline__ unsigned int pack2bf(float x, float y) {
    return (unsigned int)f2bfh(x) | ((unsigned int)f2bfh(y) << 16);
}

// ---------------------------------------------------------------------------
// Tiled bf16 WMMA GEMM:  C[M x Nmat] = A[M x Kdim] * B  (+ epilogue)
//   EPI 0: C = relu(acc + bias[n])           -> bf16 (ushort) store
//   EPI 1: C = acc + bias[n]                 -> fp32 store
//   EPI 2: C = 2*acc - znorm[m] - cbnorm[n]  -> fp32 store  (= -distances)
//   A_BF16:       A is bf16 in global; staged with global_load_async_to_lds
//   B_ROWMAJOR_N: B is stored [Nmat x Kdim] row-major (codebook), else [Kdim x Nmat]
// Block: 256 threads = 8 waves (2x4), macro tile 128x128, K-step 32.
// ---------------------------------------------------------------------------
template <int EPI, bool A_BF16, bool B_ROWMAJOR_N>
__global__ __launch_bounds__(256)
void gemm_wmma(const void* __restrict__ Ap, const void* __restrict__ Bp,
               const float* __restrict__ bias, const float* __restrict__ znorm,
               const float* __restrict__ cbnorm, void* __restrict__ Cp,
               int M, int Nmat, int Kdim)
{
    constexpr int LDT = 34;                      // padded LDS stride (halves)
    __shared__ unsigned short As[128 * LDT];     // A tile  [m][k]
    __shared__ unsigned short Bs[128 * LDT];     // B tile  [n][k]

    const int tid  = threadIdx.x;
    const int lane = tid & 31;
    const int wave = tid >> 5;
    const int wm   = wave >> 2;     // 0..1  (64 rows each)
    const int wn   = wave & 3;      // 0..3  (32 cols each)
    const int l16  = lane & 15;
    const int lhi  = lane >> 4;     // 0/1

    const int gm0 = blockIdx.y * 128;
    const int gn0 = blockIdx.x * 128;

    v8f acc[4][2];
    #pragma unroll
    for (int i = 0; i < 4; ++i)
        #pragma unroll
        for (int j = 0; j < 2; ++j)
            #pragma unroll
            for (int e = 0; e < 8; ++e)
                acc[i][j][e] = 0.0f;

    for (int k0 = 0; k0 < Kdim; k0 += 32) {
        // ---------------- stage A tile (128 rows x 32 k) -------------------
        {
            const int r  = tid >> 1;             // 0..127
            const int c0 = (tid & 1) << 4;       // 0 or 16
            if constexpr (A_BF16) {
                // CDNA5 async copy: global -> LDS, no VGPR round-trip.
                // VDST operand = LDS byte offset (low 32 bits of flat addr).
                const unsigned short* g = (const unsigned short*)Ap +
                    (size_t)(gm0 + r) * Kdim + k0 + c0;
                unsigned int lds0 = (unsigned int)(uintptr_t)&As[r * LDT + c0];
                asm volatile("global_load_async_to_lds_b128 %0, %1, off"
                             :: "v"(lds0), "v"(g) : "memory");
                asm volatile("global_load_async_to_lds_b128 %0, %1, off"
                             :: "v"(lds0 + 16u), "v"(g + 8) : "memory");
            } else {
                const float* g = (const float*)Ap + (size_t)(gm0 + r) * Kdim + k0 + c0;
                unsigned int* dst = (unsigned int*)&As[r * LDT + c0];
                #pragma unroll
                for (int q = 0; q < 4; ++q) {
                    float4 f = ((const float4*)g)[q];
                    dst[q*2+0] = pack2bf(f.x, f.y);
                    dst[q*2+1] = pack2bf(f.z, f.w);
                }
            }
        }
        // ---------------- stage B tile -> Bs[n][k] -------------------------
        if constexpr (B_ROWMAJOR_N) {
            const int n  = tid >> 1;
            const int c0 = (tid & 1) << 4;
            const float* g = (const float*)Bp + (size_t)(gn0 + n) * Kdim + k0 + c0;
            unsigned int* dst = (unsigned int*)&Bs[n * LDT + c0];
            #pragma unroll
            for (int q = 0; q < 4; ++q) {
                float4 f = ((const float4*)g)[q];
                dst[q*2+0] = pack2bf(f.x, f.y);
                dst[q*2+1] = pack2bf(f.z, f.w);
            }
        } else {
            const int kk = tid >> 3;             // 0..31
            const int n0 = (tid & 7) << 4;       // 0..112
            const float* g = (const float*)Bp + (size_t)(k0 + kk) * Nmat + gn0 + n0;
            #pragma unroll
            for (int q = 0; q < 4; ++q) {
                float4 f = ((const float4*)g)[q];
                Bs[(n0 + q*4 + 0) * LDT + kk] = f2bfh(f.x);
                Bs[(n0 + q*4 + 1) * LDT + kk] = f2bfh(f.y);
                Bs[(n0 + q*4 + 2) * LDT + kk] = f2bfh(f.z);
                Bs[(n0 + q*4 + 3) * LDT + kk] = f2bfh(f.w);
            }
        }
        if constexpr (A_BF16) {
            // async LDS writes are tracked by ASYNCcnt, not DScnt: drain
            // before the workgroup barrier so all waves see the tile.
            asm volatile("s_wait_asynccnt 0" ::: "memory");
        }
        __syncthreads();

        // prefetch next A K-tile into cache hierarchy (global_prefetch)
        if (k0 + 32 < Kdim) {
            const size_t esz = A_BF16 ? 2u : 4u;
            const char* pA = (const char*)Ap +
                ((size_t)(gm0 + (tid >> 1)) * Kdim + (size_t)(k0 + 32) +
                 (size_t)((tid & 1) << 4)) * esz;
            __builtin_prefetch(pA, 0, 1);
        }

        // ---------------- WMMA compute -------------------------------------
        // B fragment: lane holds col n=l16; 16 consecutive K halves starting
        // at lhi*16 (contiguous k order, same as the A fragment).
        v16bf bfrag[2];
        #pragma unroll
        for (int j = 0; j < 2; ++j) {
            const int n = wn * 32 + j * 16 + l16;
            const unsigned int* src = (const unsigned int*)&Bs[n * LDT + lhi * 16];
            unsigned int* du = (unsigned int*)&bfrag[j];
            #pragma unroll
            for (int v = 0; v < 8; ++v) du[v] = src[v];
        }
        // A fragment: lane holds row m=l16; K = lhi*16 + v*2 + c
        #pragma unroll
        for (int i = 0; i < 4; ++i) {
            const int m = wm * 64 + i * 16 + l16;
            const unsigned short* arow = &As[m * LDT + lhi * 16];
            v16bf afrag;
            unsigned int* au = (unsigned int*)&afrag;
            #pragma unroll
            for (int v = 0; v < 8; ++v)
                au[v] = *(const unsigned int*)&arow[v << 1];
            #pragma unroll
            for (int j = 0; j < 2; ++j)
                acc[i][j] = __builtin_amdgcn_wmma_f32_16x16x32_bf16(
                    false, afrag, false, bfrag[j], (short)0, acc[i][j],
                    false, false);
        }
        __syncthreads();
    }

    // ---------------- epilogue ---------------------------------------------
    #pragma unroll
    for (int i = 0; i < 4; ++i) {
        #pragma unroll
        for (int j = 0; j < 2; ++j) {
            const int n     = gn0 + wn * 32 + j * 16 + l16;
            const int mbase = gm0 + wm * 64 + i * 16 + (lhi << 3);
            if constexpr (EPI == 0) {
                const float b = bias[n];
                unsigned short* C = (unsigned short*)Cp;
                #pragma unroll
                for (int e = 0; e < 8; ++e) {
                    float v = acc[i][j][e] + b;
                    v = fmaxf(v, 0.0f);
                    C[(size_t)(mbase + e) * Nmat + n] = f2bfh(v);
                }
            } else if constexpr (EPI == 1) {
                const float b = bias[n];
                float* C = (float*)Cp;
                #pragma unroll
                for (int e = 0; e < 8; ++e)
                    C[(size_t)(mbase + e) * Nmat + n] = acc[i][j][e] + b;
            } else {
                const float cn = cbnorm[n];
                float* C = (float*)Cp;
                #pragma unroll
                for (int e = 0; e < 8; ++e) {
                    const float zn = znorm[mbase + e];
                    C[(size_t)(mbase + e) * Nmat + n] = 2.0f * acc[i][j][e] - zn - cn;
                }
            }
        }
    }
}

// ---------------------------------------------------------------------------
// Row squared-norm: out[row] = sum(A[row][:]^2)
// ---------------------------------------------------------------------------
__global__ __launch_bounds__(128)
void rownorm_kernel(const float* __restrict__ A, float* __restrict__ out, int cols)
{
    __shared__ float red[128];
    const int row = blockIdx.x;
    const float* a = A + (size_t)row * cols;
    float s = 0.0f;
    for (int c = threadIdx.x; c < cols; c += 128) { float v = a[c]; s += v * v; }
    red[threadIdx.x] = s;
    __syncthreads();
    for (int off = 64; off > 0; off >>= 1) {
        if (threadIdx.x < off) red[threadIdx.x] += red[threadIdx.x + off];
        __syncthreads();
    }
    if (threadIdx.x == 0) out[row] = red[0];
}

__global__ void init_kernel(float* hist, float* sse)
{
    const int t = blockIdx.x * blockDim.x + threadIdx.x;
    if (t < K_LAT) hist[t] = 0.0f;
    if (t == 0) *sse = 0.0f;
}

// ---------------------------------------------------------------------------
// Per-row Gumbel-argmax sample; writes one-hot encodings + quantized_st,
// accumulates SSE (sum of (q - z)^2) and the index histogram.
// ---------------------------------------------------------------------------
__global__ __launch_bounds__(256)
void sample_kernel(const float* __restrict__ negd, const float* __restrict__ z,
                   const float* __restrict__ cb, float* __restrict__ enc,
                   float* __restrict__ qst, float* __restrict__ hist,
                   float* __restrict__ sse)
{
    __shared__ float sVal[256];
    __shared__ int   sIdx[256];
    __shared__ int   sWin;
    const int row = blockIdx.x;
    const int tid = threadIdx.x;
    const float* nd = negd + (size_t)row * K_LAT;

    float best = -3.4e38f; int bidx = 0;
    for (int k = tid; k < K_LAT; k += 256) {
        unsigned int h = (unsigned int)(row * 1031 + k) * 0x9E3779B9u + 42u;
        h ^= h >> 16; h *= 0x7FEB352Du; h ^= h >> 15; h *= 0x846CA68Bu; h ^= h >> 16;
        const float u = ((float)(h >> 8) + 0.5f) * (1.0f / 16777216.0f);
        const float g = -logf(-logf(u));                  // Gumbel(0,1)
        const float score = nd[k] * INV_TEMP + g;
        if (score > best) { best = score; bidx = k; }
    }
    sVal[tid] = best; sIdx[tid] = bidx;
    __syncthreads();
    for (int off = 128; off > 0; off >>= 1) {
        if (tid < off && sVal[tid + off] > sVal[tid]) {
            sVal[tid] = sVal[tid + off]; sIdx[tid] = sIdx[tid + off];
        }
        __syncthreads();
    }
    if (tid == 0) sWin = sIdx[0];
    __syncthreads();
    const int idx = sWin;

    for (int k = tid; k < K_LAT; k += 256)
        enc[(size_t)row * K_LAT + k] = (k == idx) ? 1.0f : 0.0f;

    const float* q  = cb + (size_t)idx * F_DIM;
    const float* zr = z  + (size_t)row * F_DIM;
    float a = 0.0f;
    for (int f = tid; f < F_DIM; f += 256) {
        const float qv = q[f];
        qst[(size_t)row * F_DIM + f] = qv;                // q_st == q numerically
        const float d = qv - zr[f];
        a += d * d;
    }
    sVal[tid] = a;
    __syncthreads();
    for (int off = 128; off > 0; off >>= 1) {
        if (tid < off) sVal[tid] += sVal[tid + off];
        __syncthreads();
    }
    if (tid == 0) {
        atomicAdd(sse, sVal[0]);
        atomicAdd(&hist[idx], 1.0f);
    }
}

__global__ __launch_bounds__(1024)
void finalize_kernel(const float* __restrict__ hist, const float* __restrict__ sse,
                     float* loss_out, float* perp_out)
{
    __shared__ float red[1024];
    const int t = threadIdx.x;
    const float p = hist[t] * (1.0f / (float)N_ROWS);
    red[t] = p * logf(p + 1e-10f);
    __syncthreads();
    for (int off = 512; off > 0; off >>= 1) {
        if (t < off) red[t] += red[t + off];
        __syncthreads();
    }
    if (t == 0) {
        *perp_out = expf(-red[0]);
        // loss = q_latent + 1.0 * e_latent == 2 * mean((q - z)^2)
        *loss_out = 2.0f * (*sse) / ((float)N_ROWS * (float)F_DIM);
    }
}

// ---------------------------------------------------------------------------
extern "C" void kernel_launch(void* const* d_in, const int* in_sizes, int n_in,
                              void* d_out, int out_size, void* d_ws, size_t ws_size,
                              hipStream_t stream)
{
    const float* X  = (const float*)d_in[0];   // [N, D]
    const float* W1 = (const float*)d_in[1];   // [D, H]
    const float* b1 = (const float*)d_in[2];   // [H]
    const float* W2 = (const float*)d_in[3];   // [H, F]
    const float* b2 = (const float*)d_in[4];   // [F]
    const float* cb = (const float*)d_in[5];   // [K, F]
    float* out = (float*)d_out;

    // workspace carve-out
    char* ws = (char*)d_ws;
    float*          z     = (float*)ws;          ws += (size_t)N_ROWS * F_DIM * 4;
    unsigned short* Hbf   = (unsigned short*)ws; ws += (size_t)N_ROWS * H_DIM * 2;
    float*          znorm = (float*)ws;          ws += (size_t)N_ROWS * 4;
    float*          cbn   = (float*)ws;          ws += (size_t)K_LAT * 4;
    float*          hist  = (float*)ws;          ws += (size_t)K_LAT * 4;
    float*          sse   = (float*)ws;          ws += 256;

    // d_out layout: [loss(1), quantized_st(N*F), perplexity(1), encodings(N*K), -dist(N*K)]
    const size_t OFF_QST  = 1;
    const size_t OFF_PERP = 1 + (size_t)N_ROWS * F_DIM;
    const size_t OFF_ENC  = OFF_PERP + 1;
    const size_t OFF_NEGD = OFF_ENC + (size_t)N_ROWS * K_LAT;

    init_kernel<<<4, 256, 0, stream>>>(hist, sse);
    rownorm_kernel<<<K_LAT, 128, 0, stream>>>(cb, cbn, F_DIM);

    // H = relu(X @ W1 + b1)  -> bf16
    gemm_wmma<0, false, false><<<dim3(H_DIM / 128, N_ROWS / 128), 256, 0, stream>>>(
        X, W1, b1, nullptr, nullptr, Hbf, N_ROWS, H_DIM, D_IN);
    // z = H @ W2 + b2  -> fp32  (A tile staged via global_load_async_to_lds)
    gemm_wmma<1, true, false><<<dim3(F_DIM / 128, N_ROWS / 128), 256, 0, stream>>>(
        Hbf, W2, b2, nullptr, nullptr, z, N_ROWS, F_DIM, H_DIM);
    rownorm_kernel<<<N_ROWS, 128, 0, stream>>>(z, znorm, F_DIM);
    // -dist = 2*z@cb^T - |z|^2 - |cb|^2  -> straight into d_out
    gemm_wmma<2, false, true><<<dim3(K_LAT / 128, N_ROWS / 128), 256, 0, stream>>>(
        z, cb, nullptr, znorm, cbn, out + OFF_NEGD, N_ROWS, K_LAT, F_DIM);

    sample_kernel<<<N_ROWS, 256, 0, stream>>>(
        out + OFF_NEGD, z, cb, out + OFF_ENC, out + OFF_QST, hist, sse);
    finalize_kernel<<<1, 1024, 0, stream>>>(hist, sse, out, out + OFF_PERP);
}
